// MlpMixer_18966575579742
// MI455X (gfx1250) — matrix operand order repaired and checked by reading, not compiled, not run on hardware
//
#include <hip/hip_runtime.h>
#include <math.h>

typedef __attribute__((ext_vector_type(16))) __bf16 v16bf;
typedef __attribute__((ext_vector_type(8)))  __bf16 v8bf;
typedef __attribute__((ext_vector_type(8)))  float  v8f;

#define Jdim  26
#define Kdim  128
#define Hdim  512
#define MROWS 2048   // B*I = 128*16

// ---------------- weight prep: fp32 -> bf16 in WMMA B-fragment order -------
// B-fragment layout (16-bit B, 32x16 KxN): element e of lane L holds
//   K = 8*(L>>4) + (e<8 ? e : 16 + (e-8)) ,  N = (L&15)
// w1 prepped: [sel(2)][j][nt(32)][kk(4)][lane(32)][e(16)]
__global__ void prep_w1(const float* __restrict__ w1, __bf16* __restrict__ dst) {
  int idx = blockIdx.x * blockDim.x + threadIdx.x;
  const int TOT = 2 * Jdim * 65536;           // 16*32*4*32 = 65536 per (sel,j)
  if (idx >= TOT) return;
  int e  = idx & 15;
  int L  = (idx >> 4) & 31;
  int kk = (idx >> 9) & 3;
  int nt = (idx >> 11) & 31;
  int sj = idx >> 16;
  int j  = sj % Jdim;
  int sel = sj / Jdim;
  int k = kk * 32 + ((L >> 4) << 3) + e + ((e >> 3) << 3);      // 0..127
  int h = nt * 16 + (L & 15);                 // 0..511
  dst[idx] = (__bf16)w1[(((size_t)sel * Jdim + j) * Kdim + k) * Hdim + h];
}

// w2 prepped: [sel(3)][j][nt2(8)][kk2(16)][lane(32)][e(16)]
//   sel 0 = w2_real, sel 1 = w2_imag, sel 2 = w2_real + w2_imag
__global__ void prep_w2(const float* __restrict__ w2, __bf16* __restrict__ dst) {
  int idx = blockIdx.x * blockDim.x + threadIdx.x;
  const int TOT = 3 * Jdim * 65536;           // 16*32*16*8 = 65536 per (sel,j)
  if (idx >= TOT) return;
  int e   = idx & 15;
  int L   = (idx >> 4) & 31;
  int kk2 = (idx >> 9) & 15;
  int nt2 = (idx >> 13) & 7;
  int sj  = idx >> 16;
  int j   = sj % Jdim;
  int sel = sj / Jdim;
  int hk = kk2 * 32 + ((L >> 4) << 3) + e + ((e >> 3) << 3);    // 0..511 (hidden dim = WMMA K)
  int n  = nt2 * 16 + (L & 15);               // 0..127 (output dim = WMMA N)
  size_t o0 = (((size_t)0 * Jdim + j) * Hdim + hk) * Kdim + n;
  size_t o1 = (((size_t)1 * Jdim + j) * Hdim + hk) * Kdim + n;
  float v = (sel == 0) ? w2[o0] : (sel == 1) ? w2[o1] : (w2[o0] + w2[o1]);
  dst[idx] = (__bf16)v;
}

// ---------------- branchless exact-GELU (erf) -------------------------------
// Abramowitz–Stegun 7.1.26 erf approximation, |abs err| < 1.5e-7:
// straight-line VALU + TRANS ops (v_rcp_f32, v_exp_f32), no exec branching,
// co-executes with the XDL WMMA pipe.
__device__ __forceinline__ float gelu_erf(float x) {
  float z = x * 0.7071067811865476f;
  float a = fabsf(z);
  float t = __builtin_amdgcn_rcpf(__builtin_fmaf(0.3275911f, a, 1.0f));
  float p = t * (0.254829592f +
            t * (-0.284496736f +
            t * (1.421413741f +
            t * (-1.453152027f +
            t * 1.061405429f))));
  float e  = __expf(-a * a);
  float er = copysignf(__builtin_fmaf(-p, e, 1.0f), z);
  return 0.5f * x * (1.0f + er);
}

// ---------------- fused complex MLP kernel ---------------------------------
__global__ __launch_bounds__(256) void fused_mlp(
    const float* __restrict__ xr, const float* __restrict__ xi,
    const float* __restrict__ b1, const float* __restrict__ b2,
    const __bf16* __restrict__ w1p, const __bf16* __restrict__ w2p,
    float* __restrict__ out)
{
  // A-operand tiles stored in WMMA 16-bit A fragment order:
  //   element e of lane L holds row (L&15),
  //   K = base + 8*(L>>4) + (e<8 ? e : 16 + (e-8))
  __shared__ __align__(32) __bf16 sar[4][32][16];    // x_real  (K=128 -> 4 kk)
  __shared__ __align__(32) __bf16 sai[4][32][16];    // x_imag
  __shared__ __align__(32) __bf16 sh1r[16][32][16];  // gelu o1 real (H=512)
  __shared__ __align__(32) __bf16 sh1i[16][32][16];  // gelu o1 imag

  const int mt   = blockIdx.x;        // 0..127 : 16-row tile of M=2048
  const int j    = blockIdx.y;        // 0..25  : frequency group
  const int t    = threadIdx.x;
  const int lane = t & 31;
  const int wave = t >> 5;            // 8 waves

  // ---- stage x tile: fp32 global -> bf16 LDS directly in fragment order --
  // 256 threads, each produces one 8-element (16B) run:
  //   kk = t>>6, L = (t>>1)&31, half = t&1
  //   K  = kk*32 + 8*(L>>4) + 16*half + [0..8),  row = L&15
  {
    int kk   = t >> 6;
    int L    = (t >> 1) & 31;
    int half = t & 1;
    int row  = L & 15;
    int kb   = kk * 32 + ((L >> 4) << 3) + (half << 4);
    size_t g = (((size_t)(mt * 16 + row)) * Jdim + j) * Kdim + kb;
    float4 r0 = *(const float4*)(xr + g);
    float4 r1 = *(const float4*)(xr + g + 4);
    float4 i0 = *(const float4*)(xi + g);
    float4 i1 = *(const float4*)(xi + g + 4);
    v8bf pr, pi;
    pr[0] = (__bf16)r0.x; pr[1] = (__bf16)r0.y; pr[2] = (__bf16)r0.z; pr[3] = (__bf16)r0.w;
    pr[4] = (__bf16)r1.x; pr[5] = (__bf16)r1.y; pr[6] = (__bf16)r1.z; pr[7] = (__bf16)r1.w;
    pi[0] = (__bf16)i0.x; pi[1] = (__bf16)i0.y; pi[2] = (__bf16)i0.z; pi[3] = (__bf16)i0.w;
    pi[4] = (__bf16)i1.x; pi[5] = (__bf16)i1.y; pi[6] = (__bf16)i1.z; pi[7] = (__bf16)i1.w;
    *(v8bf*)&sar[kk][L][half * 8] = pr;
    *(v8bf*)&sai[kk][L][half * 8] = pi;
  }
  __syncthreads();

  // ---- layer-1 A fragments: straight 32B LDS loads, shared by all waves --
  v16bf axr[4], axi[4];
  #pragma unroll
  for (int kk = 0; kk < 4; ++kk) {
    axr[kk] = *(const v16bf*)&sar[kk][lane][0];
    axi[kk] = *(const v16bf*)&sai[kk][lane][0];
  }

  // ---- layer 1: each wave computes 4 of the 32 N-tiles of H=512 ----
  #pragma unroll
  for (int tl = 0; tl < 4; ++tl) {
    int nt = wave * 4 + tl;
    v8f aRR = {}, aII = {}, aIR = {}, aRI = {};
    const __bf16* pr = w1p + ((((size_t)0 * Jdim + j) * 32 + nt) * 4) * 512;
    const __bf16* pi = w1p + ((((size_t)1 * Jdim + j) * 32 + nt) * 4) * 512;
    #pragma unroll
    for (int kk = 0; kk < 4; ++kk) {
      v16bf br = *(const v16bf*)(pr + kk * 512 + lane * 16);
      v16bf bi = *(const v16bf*)(pi + kk * 512 + lane * 16);
      aRR = __builtin_amdgcn_wmma_f32_16x16x32_bf16(false, axr[kk], false, br, (short)0, aRR, false, false);
      aII = __builtin_amdgcn_wmma_f32_16x16x32_bf16(false, axi[kk], false, bi, (short)0, aII, false, false);
      aIR = __builtin_amdgcn_wmma_f32_16x16x32_bf16(false, axi[kk], false, br, (short)0, aIR, false, false);
      aRI = __builtin_amdgcn_wmma_f32_16x16x32_bf16(false, axr[kk], false, bi, (short)0, aRI, false, false);
    }
    // combine, bias, branchless erf-GELU; deposit bf16 hidden into LDS in
    // the layer-2 A-fragment position for hidden index h = n:
    //   kk2 = h>>5, L = m + 16*((h>>3)&1), e = (h&7) + 8*((h>>4)&1)
    int   n   = nt * 16 + (lane & 15);
    float br1 = b1[((size_t)0 * Jdim + j) * Hdim + n];
    float bi1 = b1[((size_t)1 * Jdim + j) * Hdim + n];
    int kk2 = n >> 5;
    int c   = n & 31;
    int Lg  = ((c >> 3) & 1) << 4;               // +16 if K-slot group 1
    int ew  = (c & 7) + (((c >> 4) & 1) << 3);   // element within fragment
    int mbase = (lane >> 4) * 8;                 // C layout rows
    #pragma unroll
    for (int r = 0; r < 8; ++r) {
      float vr = aRR[r] - aII[r] + br1;
      float vi = aIR[r] + aRI[r] + bi1;
      int Lw = (mbase + r) + Lg;
      sh1r[kk2][Lw][ew] = (__bf16)gelu_erf(vr);
      sh1i[kk2][Lw][ew] = (__bf16)gelu_erf(vi);
    }
  }
  __syncthreads();

  // ---- layer 2: wave owns N-tile nt2 = wave (8 x 16 = 128 outputs) ----
  {
    v8f aA = {}, aB = {}, aC = {};   // o1r*w2r, o1i*w2i, o1i*(w2r+w2i)
    const __bf16* p0 = w2p + ((((size_t)0 * Jdim + j) * 8 + wave) * 16) * 512;
    const __bf16* p1 = w2p + ((((size_t)1 * Jdim + j) * 8 + wave) * 16) * 512;
    const __bf16* p2 = w2p + ((((size_t)2 * Jdim + j) * 8 + wave) * 16) * 512;
    #pragma unroll 4
    for (int kk = 0; kk < 16; ++kk) {
      v16bf ar = *(const v16bf*)&sh1r[kk][lane][0];
      v16bf ai = *(const v16bf*)&sh1i[kk][lane][0];
      v16bf br = *(const v16bf*)(p0 + kk * 512 + lane * 16);
      v16bf bi = *(const v16bf*)(p1 + kk * 512 + lane * 16);
      v16bf bs = *(const v16bf*)(p2 + kk * 512 + lane * 16);
      aA = __builtin_amdgcn_wmma_f32_16x16x32_bf16(false, ar, false, br, (short)0, aA, false, false);
      aB = __builtin_amdgcn_wmma_f32_16x16x32_bf16(false, ai, false, bi, (short)0, aB, false, false);
      aC = __builtin_amdgcn_wmma_f32_16x16x32_bf16(false, ai, false, bs, (short)0, aC, false, false);
    }
    int   n   = wave * 16 + (lane & 15);
    float br2 = b2[((size_t)0 * Jdim + j) * Kdim + n];
    float bi2 = b2[((size_t)1 * Jdim + j) * Kdim + n];
    int mbase = (lane >> 4) * 8;
    #pragma unroll
    for (int r = 0; r < 8; ++r) {
      int grow = mt * 16 + mbase + r;
      size_t off = 2 * ((((size_t)grow) * Jdim + j) * Kdim + n);
      float2 o;
      o.x = aA[r] - aB[r] + br2;   // o2_real
      o.y = aC[r] + bi2;           // o2_imag (reference quirk: o1i*(w2r+w2i))
      *(float2*)(out + off) = o;
    }
  }
}

// ---------------- host entry -----------------------------------------------
extern "C" void kernel_launch(void* const* d_in, const int* in_sizes, int n_in,
                              void* d_out, int out_size, void* d_ws, size_t ws_size,
                              hipStream_t stream) {
  const float* xr = (const float*)d_in[0];
  const float* xi = (const float*)d_in[1];
  const float* w1 = (const float*)d_in[2];
  const float* b1 = (const float*)d_in[3];
  const float* w2 = (const float*)d_in[4];
  const float* b2 = (const float*)d_in[5];
  float* out = (float*)d_out;

  const size_t W1E = (size_t)2 * Jdim * 32 * 4 * 512;   // 3,407,872 bf16
  const size_t W2E = (size_t)3 * Jdim * 8 * 16 * 512;   // 5,111,808 bf16
  __bf16* w1p = (__bf16*)d_ws;
  __bf16* w2p = (__bf16*)((char*)d_ws + W1E * 2);

  prep_w1<<<(int)((W1E + 255) / 256), 256, 0, stream>>>(w1, w1p);
  prep_w2<<<(int)((W2E + 255) / 256), 256, 0, stream>>>(w2, w2p);

  dim3 grid(MROWS / 16, Jdim);
  fused_mlp<<<grid, 256, 0, stream>>>(xr, xi, b1, b2, w1p, w2p, out);
}
